// proj_loss_35974646072032
// MI455X (gfx1250) — matrix operand order repaired and checked by reading, not compiled, not run on hardware
//
#include <hip/hip_runtime.h>

#define N_PTS 4096
#define CHN   768
#define KSPEC 128
#define KNN   16
#define NITER 12
#define JSWEEPS 6
#define LAMBDA_W 100.0f

typedef __attribute__((ext_vector_type(16))) __bf16 v16bf;
typedef __attribute__((ext_vector_type(8)))  float  v8f;
typedef __attribute__((ext_vector_type(4)))  unsigned int u32x4;
typedef __attribute__((ext_vector_type(8)))  int i32x8;
typedef __attribute__((ext_vector_type(4)))  int i32x4;

union ABPack { unsigned int u[8]; v16bf v; };

__device__ __forceinline__ unsigned short f2bf(float f) {
  unsigned int u = __float_as_uint(f);
  u += 0x7FFFu + ((u >> 16) & 1u);   // round-to-nearest-even
  return (unsigned short)(u >> 16);
}

// ---------------------------------------------------------------------------
// TDM 2-D tile load: global (row-major, bf16) -> LDS, per cdna5 D# layout.
//   tile = tile_d1 rows x 32 elements, row stride = stride0 elements.
// Group0: [1:0]=count=1, [63:32]=lds_addr, [120:64]=global_addr, [127:126]=2
// Group1: [17:16]=data_size(1=2B), [79:48]=tensor_dim0, [111:80]=tensor_dim1,
//         [127:112]=tile_dim0, [143:128]=tile_dim1, [207:160]=dim0_stride
// ---------------------------------------------------------------------------
__device__ __forceinline__ void tdm_load_2d(unsigned long long gaddr,
                                            unsigned int lds_off,
                                            unsigned int tensor_d0,
                                            unsigned int tensor_d1,
                                            unsigned long long stride0,
                                            unsigned int tile_d1) {
  u32x4 g0;
  g0[0] = 1u;                                            // count=1 (valid, user)
  g0[1] = lds_off;                                       // LDS byte address
  g0[2] = (unsigned int)(gaddr & 0xFFFFFFFFull);         // global_addr[31:0]
  g0[3] = (unsigned int)((gaddr >> 32) & 0x1FFFFFFull)   // global_addr[56:32]
          | (2u << 30);                                  // type=2 ("image")
  i32x8 g1;
  g1[0] = (int)(1u << 16);                               // data_size=1 -> 2 bytes
  g1[1] = (int)((tensor_d0 & 0xFFFFu) << 16);            // dim0[15:0] @ [63:48]
  g1[2] = (int)((tensor_d0 >> 16) |                      // dim0[31:16]
                ((tensor_d1 & 0xFFFFu) << 16));          // dim1[15:0]
  g1[3] = (int)((tensor_d1 >> 16) | (32u << 16));        // dim1[31:16], tile_d0=32
  g1[4] = (int)(tile_d1 & 0xFFFFu);                      // tile_dim1, tile_dim2=0
  g1[5] = (int)(stride0 & 0xFFFFFFFFull);                // dim0_stride[31:0]
  g1[6] = (int)((stride0 >> 32) & 0xFFFFull);            // dim0_stride[47:32]
  g1[7] = 0;
  i32x4 z4 = {0, 0, 0, 0};
#if defined(__clang_major__) && (__clang_major__ >= 23)
  i32x8 z8 = {0, 0, 0, 0, 0, 0, 0, 0};
  __builtin_amdgcn_tensor_load_to_lds(g0, g1, z4, z4, z8, 0);
#else
  __builtin_amdgcn_tensor_load_to_lds(g0, g1, z4, z4, 0);
#endif
}

// ---------------------------------------------------------------------------
// TDM-fed WMMA GEMM: C[M,N] = X[M,K] * Y[N,K]^T.
// One 256-thread block (8 waves) per 128x128 output macro-tile. Per K-step of
// 32, wave 0 TDM-loads a 128x32 X panel and 128x32 Y panel into LDS, waits on
// TENSORcnt, then each wave computes a 2x4 grid of 16x16 WMMA tiles from LDS.
// Requires M%128==0, N%128==0, K%32==0.
// ---------------------------------------------------------------------------
__global__ void k_gemm_tdm_bf16(const unsigned short* __restrict__ Xh,
                                const unsigned short* __restrict__ Yh,
                                float* __restrict__ C,
                                int Mb, int Nb, int K, int ldx, int ldy, int ldc,
                                int Mtot, int Ntot) {
  __shared__ __align__(16) unsigned short s_pan[2 * 128 * 32]; // X @0, Y @8KB
  int tid = threadIdx.x;
  int lane = tid & 31;
  int w = tid >> 5;
  int bm = blockIdx.x / Nb, bn = blockIdx.x % Nb;
  int khalf = lane >> 4;
  int l15 = lane & 15;
  int mpair = w >> 1;   // m-tiles {2*mpair, 2*mpair+1}
  int npair = w & 1;    // n-tiles {4*npair .. 4*npair+3}
  const unsigned int* Xs = (const unsigned int*)s_pan;              // 128x16 dw
  const unsigned int* Ys = (const unsigned int*)(s_pan + 128 * 32); // 128x16 dw
  unsigned long long xbase = (unsigned long long)(uintptr_t)Xh +
                             (((unsigned long long)(bm * 128) * (unsigned)ldx) << 1);
  unsigned long long ybase = (unsigned long long)(uintptr_t)Yh +
                             (((unsigned long long)(bn * 128) * (unsigned)ldy) << 1);
  v8f acc[2][4] = {};
  for (int k0 = 0; k0 < K; k0 += 32) {
    if (w == 0) {
      tdm_load_2d(xbase + ((unsigned long long)k0 << 1), 0u,
                  (unsigned)ldx, (unsigned)Mtot, (unsigned long long)ldx, 128u);
      tdm_load_2d(ybase + ((unsigned long long)k0 << 1), 128u * 32u * 2u,
                  (unsigned)ldy, (unsigned)Ntot, (unsigned long long)ldy, 128u);
      __builtin_amdgcn_s_wait_tensorcnt(0);
    }
    __syncthreads();
    ABPack a[2], b[4];
#pragma unroll
    for (int t = 0; t < 2; ++t) {
      int mr = (2 * mpair + t) * 16 + l15;
#pragma unroll
      for (int r = 0; r < 8; ++r)
        a[t].u[r] = Xs[mr * 16 + (r < 4 ? 0 : 8) + khalf * 4 + (r & 3)];
    }
#pragma unroll
    for (int t = 0; t < 4; ++t) {
      int nr = (4 * npair + t) * 16 + l15;
#pragma unroll
      for (int r = 0; r < 8; ++r)
        b[t].u[r] = Ys[nr * 16 + (r < 4 ? 0 : 8) + khalf * 4 + (r & 3)];
    }
#pragma unroll
    for (int i = 0; i < 2; ++i)
#pragma unroll
      for (int j = 0; j < 4; ++j)
        acc[i][j] = __builtin_amdgcn_wmma_f32_16x16x32_bf16(
            false, a[i].v, false, b[j].v, (short)0, acc[i][j], false, false);
    __syncthreads();
  }
#pragma unroll
  for (int i = 0; i < 2; ++i) {
#pragma unroll
    for (int j = 0; j < 4; ++j) {
      int ncol = bn * 128 + (4 * npair + j) * 16 + l15;
#pragma unroll
      for (int r = 0; r < 8; ++r) {
        int mr = bm * 128 + (2 * mpair + i) * 16 + r + 8 * khalf;
        C[(size_t)mr * ldc + ncol] = acc[i][j][r];
      }
    }
  }
}

// ---------------------------------------------------------------------------
// Direct-load WMMA GEMM: C[M,N] = X[M,K] * Y[N,K]^T (one wave per 16x16 tile)
// ---------------------------------------------------------------------------
__global__ void k_gemm_xyT_bf16(const unsigned short* __restrict__ Xh,
                                const unsigned short* __restrict__ Yh,
                                float* __restrict__ C,
                                int Mt, int Nt, int K, int ldx, int ldy, int ldc) {
  int lane = threadIdx.x & 31;
  int wave = blockIdx.x * (blockDim.x >> 5) + (threadIdx.x >> 5);
  if (wave >= Mt * Nt) return;
  int tm = wave / Nt, tn = wave % Nt;
  int khalf = lane >> 4;
  int mrow = tm * 16 + (lane & 15);
  int nrow = tn * 16 + (lane & 15);
  const unsigned int* xrow = (const unsigned int*)(Xh + (size_t)mrow * ldx);
  const unsigned int* yrow = (const unsigned int*)(Yh + (size_t)nrow * ldy);
  v8f acc = {};
  for (int k0 = 0; k0 < K; k0 += 32) {
    ABPack a, b;
    int kd = k0 >> 1;
#pragma unroll
    for (int r = 0; r < 8; ++r) {
      int ad = (r < 4 ? 0 : 8) + khalf * 4 + (r & 3);   // A 16x32 bf16 layout
      a.u[r] = xrow[kd + ad];
      int bd = (r < 4 ? 0 : 8) + khalf * 4 + (r & 3);   // B 32x16 bf16 layout
      b.u[r] = yrow[kd + bd];
    }
    if (k0 + 32 < K) {
      __builtin_prefetch(xrow + kd + 16, 0, 1);
      __builtin_prefetch(yrow + kd + 16, 0, 1);
    }
    acc = __builtin_amdgcn_wmma_f32_16x16x32_bf16(false, a.v, false, b.v,
                                                  (short)0, acc, false, false);
  }
  int ncol = tn * 16 + (lane & 15);
#pragma unroll
  for (int r = 0; r < 8; ++r) {
    int mr = tm * 16 + r + 8 * khalf;                   // C f32 16x16 layout
    C[(size_t)mr * ldc + ncol] = acc[r];
  }
}

// ---------------------------------------------------------------------------
// Elementwise / conversion helpers
// ---------------------------------------------------------------------------
__global__ void k_zero_f32(float* p, long n) {
  long i = (long)blockIdx.x * blockDim.x + threadIdx.x;
  if (i < n) p[i] = 0.f;
}

__global__ void k_f32_to_bf16(const float* __restrict__ X,
                              unsigned short* __restrict__ Y, long n) {
  long i = (long)blockIdx.x * blockDim.x + threadIdx.x;
  if (i < n) Y[i] = f2bf(X[i]);
}

// X[rows,cols] f32 -> Y[cols,rows] bf16
__global__ void k_transpose_bf16(const float* __restrict__ X,
                                 unsigned short* __restrict__ Y,
                                 int rows, int cols) {
  long i = (long)blockIdx.x * blockDim.x + threadIdx.x;
  if (i >= (long)rows * cols) return;
  int r = (int)(i / cols), c = (int)(i % cols);
  Y[(size_t)c * rows + r] = f2bf(X[i]);
}

__global__ void k_rowsq(const float* __restrict__ X, float* sq, int cols) {
  __shared__ float red[256];
  int row = blockIdx.x;
  float s = 0.f;
  for (int c = threadIdx.x; c < cols; c += 256) {
    float v = X[(size_t)row * cols + c];
    s += v * v;
  }
  red[threadIdx.x] = s; __syncthreads();
  for (int st = 128; st; st >>= 1) {
    if (threadIdx.x < st) red[threadIdx.x] += red[threadIdx.x + st];
    __syncthreads();
  }
  if (!threadIdx.x) sq[row] = red[0];
}

// ---------------------------------------------------------------------------
// kNN: per-row iterative 16-way min selection over d2 = sqi+sqj-2G
// ---------------------------------------------------------------------------
__global__ void k_topk(const float* __restrict__ G, const float* __restrict__ sq,
                       float* knnd, int* knni, float* sumacc, int n, int kk) {
  int row = blockIdx.x, tid = threadIdx.x;
  float vals[16];
  float sr = sq[row];
#pragma unroll
  for (int t = 0; t < 16; ++t) {
    int c = tid + t * 256;
    float v = sr + sq[c] - 2.f * G[(size_t)row * n + c];
    v = fmaxf(v, 0.f);
    if (c == row) v = 1e30f;
    vals[t] = v;
  }
  __shared__ float rv[256];
  __shared__ int   ri[256];
  for (int pick = 0; pick < kk; ++pick) {
    float bv = 3e38f; int bi = 0;
#pragma unroll
    for (int t = 0; t < 16; ++t) {
      int c = tid + t * 256;
      if (vals[t] < bv) { bv = vals[t]; bi = c; }
    }
    rv[tid] = bv; ri[tid] = bi; __syncthreads();
    for (int st = 128; st; st >>= 1) {
      if (tid < st) {
        if (rv[tid + st] < rv[tid] ||
            (rv[tid + st] == rv[tid] && ri[tid + st] < ri[tid])) {
          rv[tid] = rv[tid + st]; ri[tid] = ri[tid + st];
        }
      }
      __syncthreads();
    }
    int widx = ri[0]; float wval = rv[0];
    if (tid == 0) {
      knnd[row * kk + pick] = wval;
      knni[row * kk + pick] = widx;
      atomicAdd(sumacc, wval);
    }
    if ((widx & 255) == tid) vals[widx >> 8] = 3e38f;
    __syncthreads();
  }
}

__global__ void k_scatter_w(const float* __restrict__ knnd,
                            const int* __restrict__ knni,
                            const float* __restrict__ sumacc,
                            float* W, int n, int kk) {
  int idx = blockIdx.x * blockDim.x + threadIdx.x;
  if (idx >= n * kk) return;
  int i = idx / kk;
  float sigma2 = sumacc[0] / (float)(n * kk) + 1e-12f;
  float wv = __expf(-knnd[idx] / (2.f * sigma2));
  int j = knni[idx];
  atomicAdd(&W[(size_t)i * n + j], 0.5f * wv);
  atomicAdd(&W[(size_t)j * n + i], 0.5f * wv);
}

__global__ void k_rowsum_dis(const float* __restrict__ W, float* dis, int n) {
  __shared__ float red[256];
  int row = blockIdx.x;
  float s = 0.f;
  for (int c = threadIdx.x; c < n; c += 256) s += W[(size_t)row * n + c];
  red[threadIdx.x] = s; __syncthreads();
  for (int st = 128; st; st >>= 1) {
    if (threadIdx.x < st) red[threadIdx.x] += red[threadIdx.x + st];
    __syncthreads();
  }
  if (!threadIdx.x) dis[row] = rsqrtf(red[0] + 1e-8f);
}

// M = 2I - L = I + D^{-1/2} W D^{-1/2}, stored bf16 for WMMA power iteration
__global__ void k_build_M(const float* __restrict__ W, const float* __restrict__ dis,
                          unsigned short* __restrict__ Mh, int n) {
  long idx = (long)blockIdx.x * blockDim.x + threadIdx.x;
  if (idx >= (long)n * n) return;
  int i = (int)(idx / n), j = (int)(idx % n);
  float m = ((i == j) ? 1.f : 0.f) + dis[i] * W[idx] * dis[j];
  Mh[idx] = f2bf(m);
}

__global__ void k_initV(float* V, int n) {
  int i = blockIdx.x * blockDim.x + threadIdx.x;
  if (i >= n) return;
  unsigned u = (unsigned)i * 2654435761u ^ 0x9e3779b9u;
  u ^= u >> 16; u *= 0x85ebca6bu; u ^= u >> 13; u *= 0xc2b2ae35u; u ^= u >> 16;
  V[i] = (float)(u & 0xFFFFFFu) / 8388608.0f - 1.0f;
}

// C[a,b] = sum_r X[r,a] * Y[r,b]   (A^T B, small output, block per element)
__global__ void k_atb(const float* __restrict__ X, const float* __restrict__ Y,
                      float* C, int R, int K1, int K2) {
  int o = blockIdx.x;
  int a = o / K2, b = o % K2;
  float s = 0.f;
  for (int r = threadIdx.x; r < R; r += 256)
    s += X[(size_t)r * K1 + a] * Y[(size_t)r * K2 + b];
  __shared__ float red[256];
  red[threadIdx.x] = s; __syncthreads();
  for (int st = 128; st; st >>= 1) {
    if (threadIdx.x < st) red[threadIdx.x] += red[threadIdx.x + st];
    __syncthreads();
  }
  if (!threadIdx.x) C[(size_t)a * K2 + b] = red[0];
}

// C[R,Kc] = X[R,Kin] * B[Kin,Kc], B cached in dynamic LDS
__global__ void k_gemm_sr(const float* __restrict__ X, const float* __restrict__ B,
                          float* C, int R, int Kin, int Kc) {
  extern __shared__ float s_dyn[];
  for (int t = threadIdx.x; t < Kin * Kc; t += blockDim.x) s_dyn[t] = B[t];
  __syncthreads();
  int idx = blockIdx.x * blockDim.x + threadIdx.x;
  if (idx >= R * Kc) return;
  int r = idx / Kc, c = idx % Kc;
  float s = 0.f;
#pragma unroll 4
  for (int k = 0; k < Kin; ++k) s += X[(size_t)r * Kin + k] * s_dyn[k * Kc + c];
  C[idx] = s;
}

// small C[i,j] = sum_k X[i,k] * Y[j,k]   (X Y^T, 128x128)
__global__ void k_abt_small(const float* __restrict__ X, const float* __restrict__ Y,
                            float* C) {
  int idx = blockIdx.x * blockDim.x + threadIdx.x;
  if (idx >= KSPEC * KSPEC) return;
  int i = idx / KSPEC, j = idx % KSPEC;
  float s = 0.f;
#pragma unroll 4
  for (int k = 0; k < KSPEC; ++k) s += X[i * KSPEC + k] * Y[j * KSPEC + k];
  C[idx] = s;
}

__global__ void k_scale_cols(const float* __restrict__ U, const float* __restrict__ lam,
                             float* Us) {
  int idx = blockIdx.x * blockDim.x + threadIdx.x;
  if (idx >= KSPEC * KSPEC) return;
  int j = idx % KSPEC;
  Us[idx] = U[idx] * rsqrtf(fmaxf(lam[j], 1e-12f));
}

__global__ void k_finalize_T(const float* __restrict__ T0, float* T) {
  int idx = blockIdx.x * blockDim.x + threadIdx.x;
  if (idx >= KSPEC * KSPEC) return;
  int i = idx / KSPEC, j = idx % KSPEC;
  T[idx] = ((i == j) ? 2.f : 0.f) - 0.5f * (T0[i * KSPEC + j] + T0[j * KSPEC + i]);
}

// ---------------------------------------------------------------------------
// 128x128 symmetric Jacobi eigensolver, one block of 128 threads.
// A lives in 64KB dynamic LDS; eigenvector accumulation in global scratch.
// Outputs evals ascending and Uout with matching column order.
// ---------------------------------------------------------------------------
__global__ void k_jacobi128(const float* __restrict__ S, float* evals,
                            float* Uscratch, float* Uout, float* csbuf) {
  extern __shared__ float s_dyn[];
  const int K = KSPEC;
  int tid = threadIdx.x;
  for (int j = 0; j < K; ++j)
    s_dyn[tid * K + j] = 0.5f * (S[tid * K + j] + S[j * K + tid]);
  for (int j = 0; j < K; ++j)
    Uscratch[(size_t)tid * K + j] = (tid == j) ? 1.f : 0.f;
  __syncthreads();
#pragma unroll 1
  for (int sweep = 0; sweep < JSWEEPS; ++sweep) {
#pragma unroll 1
    for (int p = 0; p < K - 1; ++p) {
#pragma unroll 1
      for (int q = p + 1; q < K; ++q) {
        if (tid == 0) {
          float app = s_dyn[p * K + p], aqq = s_dyn[q * K + q], apq = s_dyn[p * K + q];
          float c = 1.f, s = 0.f;
          if (fabsf(apq) > 1e-12f) {
            float theta = 0.5f * (aqq - app) / apq;
            float t = ((theta >= 0.f) ? 1.f : -1.f) /
                      (fabsf(theta) + sqrtf(theta * theta + 1.f));
            c = rsqrtf(1.f + t * t);
            s = t * c;
          }
          csbuf[0] = c; csbuf[1] = s;
        }
        __syncthreads();
        float c = csbuf[0], s = csbuf[1];
        float vp = s_dyn[tid * K + p], vq = s_dyn[tid * K + q];
        s_dyn[tid * K + p] = c * vp - s * vq;
        s_dyn[tid * K + q] = s * vp + c * vq;
        __syncthreads();
        float wp = s_dyn[p * K + tid], wq = s_dyn[q * K + tid];
        s_dyn[p * K + tid] = c * wp - s * wq;
        s_dyn[q * K + tid] = s * wp + c * wq;
        float up = Uscratch[(size_t)tid * K + p], uq = Uscratch[(size_t)tid * K + q];
        Uscratch[(size_t)tid * K + p] = c * up - s * uq;
        Uscratch[(size_t)tid * K + q] = s * up + c * uq;
        __syncthreads();
      }
    }
  }
  // ascending sort of diag with index permutation; reuse LDS rows 0,1
  float dd = s_dyn[tid * K + tid];
  __syncthreads();
  float* dv = s_dyn;
  int* perm = (int*)(s_dyn + K);
  dv[tid] = dd; perm[tid] = tid;
  __syncthreads();
#pragma unroll 1
  for (int it = 0; it < K; ++it) {
    int bpos = 2 * tid + (it & 1);
    if (bpos + 1 < K) {
      if (dv[bpos] > dv[bpos + 1]) {
        float tf = dv[bpos]; dv[bpos] = dv[bpos + 1]; dv[bpos + 1] = tf;
        int ti = perm[bpos]; perm[bpos] = perm[bpos + 1]; perm[bpos + 1] = ti;
      }
    }
    __syncthreads();
  }
  evals[tid] = dv[tid];
  for (int j = 0; j < K; ++j)
    Uout[(size_t)tid * K + j] = Uscratch[(size_t)tid * K + perm[j]];
}

// ---------------------------------------------------------------------------
// Resolvent mask: rows from evy (target), cols from evx (source); gamma = 0.5
// ---------------------------------------------------------------------------
__global__ void k_resolvent(const float* __restrict__ evx, const float* __restrict__ evy,
                            float* D) {
  int idx = blockIdx.x * blockDim.x + threadIdx.x;
  if (idx >= KSPEC * KSPEC) return;
  int i = idx / KSPEC, j = idx % KSPEC;
  float s = fmaxf(fmaxf(evx[KSPEC - 1], evy[KSPEC - 1]), 1e-12f);
  float g1 = sqrtf(fmaxf(evx[j] / s, 0.f));
  float g2 = sqrtf(fmaxf(evy[i] / s, 0.f));
  float re = g2 / (g2 * g2 + 1.f) - g1 / (g1 * g1 + 1.f);
  float im = 1.f / (g2 * g2 + 1.f) - 1.f / (g1 * g1 + 1.f);
  D[idx] = re * re + im * im;
}

// ---------------------------------------------------------------------------
// Per-row regularized solve: (AAt + lam*diag(D[i,:])) x = BAt[i,:]
// One block (128 threads) per system; in-LDS Gaussian elimination.
// ---------------------------------------------------------------------------
__global__ void k_fm_solve(const float* __restrict__ AAt, const float* __restrict__ D,
                           const float* __restrict__ BAt, float* Cout, float lam) {
  extern __shared__ float s_dyn[];
  const int K = KSPEC;
  float* A = s_dyn;
  float* rhs = s_dyn + K * K;
  float* x = rhs + K;
  int i = blockIdx.x, tid = threadIdx.x;
  for (int j = 0; j < K; ++j) A[tid * K + j] = AAt[tid * K + j];
  A[tid * K + tid] += lam * D[i * K + tid];
  rhs[tid] = BAt[i * K + tid];
  __syncthreads();
#pragma unroll 1
  for (int k = 0; k < K - 1; ++k) {
    if (tid > k) {
      float f = A[tid * K + k] / A[k * K + k];
      for (int c = k; c < K; ++c) A[tid * K + c] -= f * A[k * K + c];
      rhs[tid] -= f * rhs[k];
    }
    __syncthreads();
  }
#pragma unroll 1
  for (int k = K - 1; k >= 0; --k) {
    if (tid == k) x[k] = rhs[k] / A[k * K + k];
    __syncthreads();
    if (tid < k) rhs[tid] -= A[tid * K + k] * x[k];
    __syncthreads();
  }
  Cout[(size_t)i * K + tid] = x[tid];
}

// ---------------------------------------------------------------------------
// SurfMNet losses -> out[3] via atomics (out zeroed beforehand)
// ---------------------------------------------------------------------------
__global__ void k_loss(const float* __restrict__ Cxy, const float* __restrict__ Cyx,
                       const float* __restrict__ evx, const float* __restrict__ evy,
                       float* out) {
  const int K = KSPEC;
  int i = blockIdx.x, j = threadIdx.x;
  float pxy = 0.f, pyx = 0.f, oxy = 0.f, oyx = 0.f;
#pragma unroll 4
  for (int k = 0; k < K; ++k) {
    pxy += Cxy[i * K + k] * Cyx[k * K + j];
    pyx += Cyx[i * K + k] * Cxy[k * K + j];
    oxy += Cxy[k * K + i] * Cxy[k * K + j];
    oyx += Cyx[k * K + i] * Cyx[k * K + j];
  }
  float dij = (i == j) ? 1.f : 0.f;
  float bij = (pxy - dij) * (pxy - dij) + (pyx - dij) * (pyx - dij);
  float ort = (oxy - dij) * (oxy - dij) + (oyx - dij) * (oyx - dij);
  float a = Cxy[i * K + j] * (evx[j] - evy[i]);
  float b = Cyx[i * K + j] * (evy[j] - evx[i]);
  float lap = a * a + b * b;
  __shared__ float r0[KSPEC], r1[KSPEC], r2[KSPEC];
  r0[j] = bij; r1[j] = ort; r2[j] = lap;
  __syncthreads();
  for (int st = K / 2; st; st >>= 1) {
    if (j < st) { r0[j] += r0[j + st]; r1[j] += r1[j + st]; r2[j] += r2[j + st]; }
    __syncthreads();
  }
  if (j == 0) {
    atomicAdd(&out[0], r0[0]);
    atomicAdd(&out[1], r1[0]);
    atomicAdd(&out[2], r2[0]);
  }
}

// ---------------------------------------------------------------------------
extern "C" void kernel_launch(void* const* d_in, const int* in_sizes, int n_in,
                              void* d_out, int out_size, void* d_ws, size_t ws_size,
                              hipStream_t stream) {
  (void)in_sizes; (void)n_in; (void)out_size; (void)ws_size;
  const int N = N_PTS, Cc = CHN, KS = KSPEC, KK = KNN;
  const float* feat_v = (const float*)d_in[0];
  const float* feat_t = (const float*)d_in[1];
  float* out = (float*)d_out;

  char* base = (char*)d_ws;
  size_t off = 0;
  auto alloc = [&](size_t bytes) -> void* {
    void* p = base + off;
    off += (bytes + 255) & ~(size_t)255;
    return p;
  };

  float*          G     = (float*)alloc((size_t)N * N * 4);          // Gram, then W
  unsigned short* Mh    = (unsigned short*)alloc((size_t)N * N * 2); // 2I - L, bf16
  unsigned short* Fh    = (unsigned short*)alloc((size_t)N * Cc * 2);
  unsigned short* FhT   = (unsigned short*)alloc((size_t)Cc * N * 2);
  float*          sq    = (float*)alloc((size_t)N * 4);
  float*          knnd  = (float*)alloc((size_t)N * KK * 4);
  int*            knni  = (int*)alloc((size_t)N * KK * 4);
  float*          scal  = (float*)alloc(256);
  float*          dis   = (float*)alloc((size_t)N * 4);
  float*          V     = (float*)alloc((size_t)N * KS * 4);
  float*          Ybuf  = (float*)alloc((size_t)N * KS * 4);
  unsigned short* Vt    = (unsigned short*)alloc((size_t)KS * N * 2);
  float*          Smat  = (float*)alloc((size_t)KS * KS * 4);
  float*          Utmp  = (float*)alloc((size_t)KS * KS * 4);
  float*          Uo    = (float*)alloc((size_t)KS * KS * 4);
  float*          Us    = (float*)alloc((size_t)KS * KS * 4);
  float*          Borth = (float*)alloc((size_t)KS * KS * 4);
  float*          lam   = (float*)alloc((size_t)KS * 4);
  float*          T0    = (float*)alloc((size_t)KS * KS * 4);
  float*          evecs_v = (float*)alloc((size_t)N * KS * 4);
  float*          evecs_t = (float*)alloc((size_t)N * KS * 4);
  float*          evals_v = (float*)alloc((size_t)KS * 4);
  float*          evals_t = (float*)alloc((size_t)KS * 4);
  unsigned short* evTh  = (unsigned short*)alloc((size_t)KS * N * 2);
  float*          Av    = (float*)alloc((size_t)KS * Cc * 4);
  float*          At_   = (float*)alloc((size_t)KS * Cc * 4);
  unsigned short* Ahv   = (unsigned short*)alloc((size_t)KS * Cc * 2);
  unsigned short* Aht   = (unsigned short*)alloc((size_t)KS * Cc * 2);
  float*          AAt_v = (float*)alloc((size_t)KS * KS * 4);
  float*          AAt_t = (float*)alloc((size_t)KS * KS * 4);
  float*          BAt_xy = (float*)alloc((size_t)KS * KS * 4);
  float*          BAt_yx = (float*)alloc((size_t)KS * KS * 4);
  float*          Dxy   = (float*)alloc((size_t)KS * KS * 4);
  float*          Dyx   = (float*)alloc((size_t)KS * KS * 4);
  float*          Cxy   = (float*)alloc((size_t)KS * KS * 4);
  float*          Cyx   = (float*)alloc((size_t)KS * KS * 4);

  const long nNN = (long)N * N;
  const long nNC = (long)N * Cc;
  const long nNK = (long)N * KS;
  const int  smKK = KS * KS * 4;              // 64KB LDS
  const int  smFM = KS * KS * 4 + 2 * KS * 4; // GE solver LDS

  auto spectral = [&](const float* feat, float* evecs, float* evals) {
    // bf16 features + squared norms
    k_f32_to_bf16<<<(int)((nNC + 255) / 256), 256, 0, stream>>>(feat, Fh, nNC);
    k_rowsq<<<N, 256, 0, stream>>>(feat, sq, Cc);
    // Gram: G = F F^T  (TDM-staged WMMA, 128x128 macro-tiles)
    k_gemm_tdm_bf16<<<(N / 128) * (N / 128), 256, 0, stream>>>(
        Fh, Fh, G, N / 128, N / 128, Cc, Cc, Cc, N, N, N);
    // kNN
    k_zero_f32<<<1, 64, 0, stream>>>(scal, 64);
    k_topk<<<N, 256, 0, stream>>>(G, sq, knnd, knni, scal, N, KK);
    // symmetric W (reuse G storage), then M = I + D^-1/2 W D^-1/2 (bf16)
    k_zero_f32<<<(int)(nNN / 256), 256, 0, stream>>>(G, nNN);
    k_scatter_w<<<(N * KK + 255) / 256, 256, 0, stream>>>(knnd, knni, scal, G, N, KK);
    k_rowsum_dis<<<N, 256, 0, stream>>>(G, dis, N);
    k_build_M<<<(int)((nNN + 255) / 256), 256, 0, stream>>>(G, dis, Mh, N);
    // subspace iteration on M (largest of M == smallest of L)
    k_initV<<<(int)((nNK + 255) / 256), 256, 0, stream>>>(V, (int)nNK);
    for (int it = 0; it < NITER; ++it) {
      k_transpose_bf16<<<(int)((nNK + 255) / 256), 256, 0, stream>>>(V, Vt, N, KS);
      k_gemm_tdm_bf16<<<(N / 128) * (KS / 128), 256, 0, stream>>>(
          Mh, Vt, Ybuf, N / 128, KS / 128, N, N, N, KS, N, KS);
      k_atb<<<KS * KS, 256, 0, stream>>>(Ybuf, Ybuf, Smat, N, KS, KS);
      k_jacobi128<<<1, KS, smKK, stream>>>(Smat, lam, Utmp, Uo, scal + 8);
      k_scale_cols<<<(KS * KS + 255) / 256, 256, 0, stream>>>(Uo, lam, Us);
      k_abt_small<<<(KS * KS + 255) / 256, 256, 0, stream>>>(Us, Uo, Borth);
      k_gemm_sr<<<(int)((nNK + 255) / 256), 256, smKK, stream>>>(Ybuf, Borth, V,
                                                                 N, KS, KS);
    }
    // Rayleigh-Ritz on L: T = 2I - V^T (M V)
    k_transpose_bf16<<<(int)((nNK + 255) / 256), 256, 0, stream>>>(V, Vt, N, KS);
    k_gemm_tdm_bf16<<<(N / 128) * (KS / 128), 256, 0, stream>>>(
        Mh, Vt, Ybuf, N / 128, KS / 128, N, N, N, KS, N, KS);
    k_atb<<<KS * KS, 256, 0, stream>>>(V, Ybuf, T0, N, KS, KS);
    k_finalize_T<<<(KS * KS + 255) / 256, 256, 0, stream>>>(T0, Smat);
    k_jacobi128<<<1, KS, smKK, stream>>>(Smat, evals, Utmp, Uo, scal + 8);
    k_gemm_sr<<<(int)((nNK + 255) / 256), 256, smKK, stream>>>(V, Uo, evecs, N, KS, KS);
  };

  spectral(feat_v, evecs_v, evals_v);
  spectral(feat_t, evecs_t, evals_t);

  // spectral coefficients A = evecs^T @ feat  (direct WMMA, K = 4096)
  auto coeffs = [&](const float* feat, float* evecs, float* Aout, unsigned short* Ah) {
    k_transpose_bf16<<<(int)((nNK + 255) / 256), 256, 0, stream>>>(evecs, evTh, N, KS);
    k_transpose_bf16<<<(int)((nNC + 255) / 256), 256, 0, stream>>>(feat, FhT, N, Cc);
    int tilesA = (KS / 16) * (Cc / 16);
    k_gemm_xyT_bf16<<<tilesA / 8, 256, 0, stream>>>(evTh, FhT, Aout, KS / 16, Cc / 16,
                                                    N, N, N, Cc);
    k_f32_to_bf16<<<(KS * Cc + 255) / 256, 256, 0, stream>>>(Aout, Ah, (long)KS * Cc);
  };
  coeffs(feat_v, evecs_v, Av, Ahv);
  coeffs(feat_t, evecs_t, At_, Aht);

  // AAt / BAt (direct WMMA, K = 768)
  int tilesS = (KS / 16) * (KS / 16);
  k_gemm_xyT_bf16<<<tilesS / 8, 256, 0, stream>>>(Ahv, Ahv, AAt_v, KS / 16, KS / 16,
                                                  Cc, Cc, Cc, KS);
  k_gemm_xyT_bf16<<<tilesS / 8, 256, 0, stream>>>(Aht, Aht, AAt_t, KS / 16, KS / 16,
                                                  Cc, Cc, Cc, KS);
  k_gemm_xyT_bf16<<<tilesS / 8, 256, 0, stream>>>(Aht, Ahv, BAt_xy, KS / 16, KS / 16,
                                                  Cc, Cc, Cc, KS);
  k_gemm_xyT_bf16<<<tilesS / 8, 256, 0, stream>>>(Ahv, Aht, BAt_yx, KS / 16, KS / 16,
                                                  Cc, Cc, Cc, KS);

  // resolvent masks + per-row regularized solves
  k_resolvent<<<(KS * KS + 255) / 256, 256, 0, stream>>>(evals_v, evals_t, Dxy);
  k_resolvent<<<(KS * KS + 255) / 256, 256, 0, stream>>>(evals_t, evals_v, Dyx);
  k_fm_solve<<<KS, KS, smFM, stream>>>(AAt_v, Dxy, BAt_xy, Cxy, LAMBDA_W);
  k_fm_solve<<<KS, KS, smFM, stream>>>(AAt_t, Dyx, BAt_yx, Cyx, LAMBDA_W);

  // losses
  k_zero_f32<<<1, 32, 0, stream>>>(out, 3);
  k_loss<<<KS, KS, 0, stream>>>(Cxy, Cyx, evals_v, evals_t, out);
}
